// PointSwinBlock_51737176047762
// MI455X (gfx1250) — matrix operand order repaired and checked by reading, not compiled, not run on hardware
//
#include <hip/hip_runtime.h>

typedef __attribute__((ext_vector_type(16))) _Float16 v16h;
typedef __attribute__((ext_vector_type(8)))  float    v8f;

#define DIM      256
#define HEADS    8
#define HEAD_DIM 32
#define WS       64
#define MLP_DIM  1024
#define NTOK     65536
#define DISP     32

// ---------- WMMA fragment helpers (contiguous k order: k = 16*(lane/16) + i for both A and B) ----------

// A-matrix 16x32 f16: lanes 0-15 -> M=lane, K=0-15; lanes 16-31 -> K=16-31
__device__ __forceinline__ v16h load_a_frag(const _Float16* __restrict__ base, int ld) {
  const int lane = threadIdx.x & 31;
  const _Float16* row = base + (lane & 15) * ld + ((lane & 16) ? 16 : 0);
  v16h a;
#pragma unroll
  for (int i = 0; i < 16; ++i) a[i] = row[i];
  return a;
}

// B-matrix 32x16 f16 read from B^T (row-major [n][k]): lanes 0-15 hold K=0-15, lanes 16-31 K=16-31
__device__ __forceinline__ v16h load_b_frag(const _Float16* __restrict__ baseT, int ld) {
  const int lane = threadIdx.x & 31;
  const _Float16* row = baseT + (lane & 15) * ld + ((lane & 16) ? 16 : 0);
  v16h b;
#pragma unroll
  for (int i = 0; i < 16; ++i) b[i] = row[i];
  return b;
}

__device__ __forceinline__ v8f wmma16(v16h a, v16h b, v8f c) {
  return __builtin_amdgcn_wmma_f32_16x16x32_f16(false, a, false, b, (short)0, c, false, false);
}

// ---------- LayerNorm: one wave handles one 256-wide row (fp32 stats, f16 out) ----------
__device__ __forceinline__ void ln_row_to_lds(const float* __restrict__ src,
                                              const float* __restrict__ gam,
                                              const float* __restrict__ bet,
                                              _Float16* __restrict__ dst) {
  const int lane = threadIdx.x & 31;
  float v[8];
  float s = 0.f, s2 = 0.f;
#pragma unroll
  for (int i = 0; i < 8; ++i) {
    float t = src[i * 32 + lane];
    v[i] = t; s += t; s2 += t * t;
  }
#pragma unroll
  for (int m = 16; m >= 1; m >>= 1) {
    s  += __shfl_xor(s,  m, 32);
    s2 += __shfl_xor(s2, m, 32);
  }
  const float mu  = s * (1.0f / 256.0f);
  const float var = s2 * (1.0f / 256.0f) - mu * mu;
  const float rs  = rsqrtf(var + 1e-5f);
#pragma unroll
  for (int i = 0; i < 8; ++i) {
    const int c = i * 32 + lane;
    dst[c] = (_Float16)((v[i] - mu) * rs * gam[c] + bet[c]);
  }
}

// ---------- weight prep: fp32 -> f16, transposed to [n][k] for B-fragments ----------
__global__ void prep_weights(const float* __restrict__ wqkv, const float* __restrict__ wout,
                             const float* __restrict__ w1,   const float* __restrict__ w2,
                             _Float16* __restrict__ ws) {
  _Float16* wqkvT = ws;                       // [768][256]
  _Float16* woutT = wqkvT + 768 * 256;        // [256][256]
  _Float16* w1T   = woutT + 256 * 256;        // [1024][256]
  _Float16* w2T   = w1T + 1024 * 256;         // [256][1024]
  const int idx = blockIdx.x * blockDim.x + threadIdx.x;
  const int stride = gridDim.x * blockDim.x;
  for (int i = idx; i < 768 * 256; i += stride) {
    int n = i >> 8, k = i & 255;
    wqkvT[i] = (_Float16)wqkv[k * 768 + n];
  }
  for (int i = idx; i < 256 * 256; i += stride) {
    int n = i >> 8, k = i & 255;
    woutT[i] = (_Float16)wout[k * 256 + n];
  }
  for (int i = idx; i < 1024 * 256; i += stride) {
    int n = i >> 8, k = i & 255;
    w1T[i] = (_Float16)w1[k * 1024 + n];
  }
  for (int i = idx; i < 256 * 1024; i += stride) {
    int n = i >> 10, k = i & 1023;
    w2T[i] = (_Float16)w2[k * 256 + n];
  }
}

// ---------- fused LN1 + QKV + shifted-window attention + out-proj + residual ----------
__launch_bounds__(256, 1)
__global__ void attn_kernel(const float* __restrict__ x,
                            const _Float16* __restrict__ wqkvT,
                            const _Float16* __restrict__ woutT,
                            const float* __restrict__ b_out,
                            const float* __restrict__ pos_emb,
                            const float* __restrict__ ln_g,
                            const float* __restrict__ ln_b,
                            float* __restrict__ x1) {
  __shared__ _Float16 s_x[WS][264];                 // LN1(x); reused as attn_out for out-proj
  __shared__ _Float16 s_q[WS][264];                 // q, col = h*32+d
  __shared__ _Float16 s_k[WS][264];                 // k, col = h*32+d
  __shared__ _Float16 s_vT[HEADS][HEAD_DIM][72];    // v transposed: [h][d][j]
  __shared__ _Float16 s_attn[HEADS][WS][72];        // softmax(QK^T) per head
  __shared__ float    s_pe[WS][66];

  const int tid  = threadIdx.x;
  // scalarize the wave index: all tile-loop bounds become SGPRs -> scalar branches,
  // WMMAs never sit under a lane-predicated region (EXEC stays all-1s).
  const int wv   = __builtin_amdgcn_readfirstlane(tid >> 5);
  const int lane = tid & 31;
  const int bat  = blockIdx.x >> 10;                // batch (2)
  const int w    = blockIdx.x & 1023;               // window (1024)
  const size_t base_tok = (size_t)bat * NTOK;
  const int nl = lane & 15;
  const int mh = (lane & 16) ? 8 : 0;

  for (int i = tid; i < WS * WS; i += 256) s_pe[i >> 6][i & 63] = pos_emb[i];

#pragma unroll
  for (int r = 0; r < 8; ++r) {
    const int j = wv * 8 + r;
    const int g = (w * WS + j + DISP) & (NTOK - 1);
    ln_row_to_lds(x + (base_tok + g) * DIM, ln_g, ln_b, &s_x[j][0]);
  }
  __syncthreads();

  // QKV GEMM: (64x256)@(256x768), split into q / k / v tile loops (all bounds scalar,
  // no store-target branches inside the WMMA loop).
  // q tiles: t in [0,64)
  for (int t = wv; t < 64; t += 8) {
    const int mt = t & 3, nt = t >> 2;
    v8f c = {};
#pragma unroll
    for (int kk = 0; kk < 8; ++kk) {
      v16h a = load_a_frag(&s_x[mt * 16][kk * 32], 264);
      v16h b = load_b_frag(wqkvT + (size_t)(nt * 16) * DIM + kk * 32, DIM);
      c = wmma16(a, b, c);
    }
    const int m0 = mt * 16 + mh, n0 = nt * 16;
#pragma unroll
    for (int r = 0; r < 8; ++r) s_q[m0 + r][n0 + nl] = (_Float16)c[r];
  }
  // k tiles: qkv cols [256,512)
  for (int t = wv; t < 64; t += 8) {
    const int mt = t & 3, nt = (t >> 2) + 16;
    v8f c = {};
#pragma unroll
    for (int kk = 0; kk < 8; ++kk) {
      v16h a = load_a_frag(&s_x[mt * 16][kk * 32], 264);
      v16h b = load_b_frag(wqkvT + (size_t)(nt * 16) * DIM + kk * 32, DIM);
      c = wmma16(a, b, c);
    }
    const int m0 = mt * 16 + mh, n0 = (nt - 16) * 16;
#pragma unroll
    for (int r = 0; r < 8; ++r) s_k[m0 + r][n0 + nl] = (_Float16)c[r];
  }
  // v tiles: qkv cols [512,768), stored transposed per head
  for (int t = wv; t < 64; t += 8) {
    const int mt = t & 3, nt = (t >> 2) + 32;
    v8f c = {};
#pragma unroll
    for (int kk = 0; kk < 8; ++kk) {
      v16h a = load_a_frag(&s_x[mt * 16][kk * 32], 264);
      v16h b = load_b_frag(wqkvT + (size_t)(nt * 16) * DIM + kk * 32, DIM);
      c = wmma16(a, b, c);
    }
    const int m0 = mt * 16 + mh;
    const int col = (nt - 32) * 16 + nl;             // 0..255
    const int h = col >> 5, d = col & 31;
#pragma unroll
    for (int r = 0; r < 8; ++r) s_vT[h][d][m0 + r] = (_Float16)c[r];
  }
  __syncthreads();

  // one head per wave: scores, softmax (register-resident, 16-lane shuffles), attn @ V
  {
    const int h = wv;                                // scalar
    const float scale = 0.17677669529663687f;        // 1/sqrt(32)
    v8f sc[4][4];
#pragma unroll
    for (int mt = 0; mt < 4; ++mt) {
      v16h a = load_a_frag(&s_q[mt * 16][h * 32], 264);
#pragma unroll
      for (int jt = 0; jt < 4; ++jt) {
        v16h b = load_b_frag(&s_k[jt * 16][h * 32], 264);
        v8f c = {};
        sc[mt][jt] = wmma16(a, b, c);
      }
    }
#pragma unroll
    for (int mt = 0; mt < 4; ++mt) {
#pragma unroll
      for (int r = 0; r < 8; ++r) {
        const int i = mt * 16 + mh + r;
        float vals[4];
        float mx = -3.0e38f;
#pragma unroll
        for (int jt = 0; jt < 4; ++jt) {
          float t = sc[mt][jt][r] * scale + s_pe[i][jt * 16 + nl];
          vals[jt] = t;
          mx = fmaxf(mx, t);
        }
#pragma unroll
        for (int m = 8; m >= 1; m >>= 1) mx = fmaxf(mx, __shfl_xor(mx, m, 32));
        float sum = 0.f;
#pragma unroll
        for (int jt = 0; jt < 4; ++jt) { vals[jt] = __expf(vals[jt] - mx); sum += vals[jt]; }
#pragma unroll
        for (int m = 8; m >= 1; m >>= 1) sum += __shfl_xor(sum, m, 32);
        const float inv = 1.0f / sum;
#pragma unroll
        for (int jt = 0; jt < 4; ++jt) s_attn[h][i][jt * 16 + nl] = (_Float16)(vals[jt] * inv);
      }
    }
    // out_h = attn (64x64) @ v (64x32): tiles 4 x 2, K = 2 x 32
#pragma unroll
    for (int mt = 0; mt < 4; ++mt) {
#pragma unroll
      for (int dt = 0; dt < 2; ++dt) {
        v8f c = {};
#pragma unroll
        for (int kk = 0; kk < 2; ++kk) {
          v16h a = load_a_frag(&s_attn[h][mt * 16][kk * 32], 72);
          v16h b = load_b_frag(&s_vT[h][dt * 16][kk * 32], 72);
          c = wmma16(a, b, c);
        }
        const int col = h * 32 + dt * 16 + nl;
        const int m0 = mt * 16 + mh;
#pragma unroll
        for (int r = 0; r < 8; ++r) s_x[m0 + r][col] = (_Float16)c[r];
      }
    }
  }
  __syncthreads();

  // out projection (64x256)@(256x256) + b_out + residual, undo shift on store
  for (int t = wv; t < 64; t += 8) {
    const int mt = t & 3, nt = t >> 2;
    v8f c = {};
#pragma unroll
    for (int kk = 0; kk < 8; ++kk) {
      v16h a = load_a_frag(&s_x[mt * 16][kk * 32], 264);
      v16h b = load_b_frag(woutT + (size_t)(nt * 16) * DIM + kk * 32, DIM);
      c = wmma16(a, b, c);
    }
    const int n = nt * 16 + nl;
    const int m0 = mt * 16 + mh;
    const float bo = b_out[n];
#pragma unroll
    for (int r = 0; r < 8; ++r) {
      const int j = m0 + r;
      const int g = (w * WS + j + DISP) & (NTOK - 1);
      const size_t idx = (base_tok + g) * DIM + n;
      x1[idx] = x[idx] + bo + c[r];
    }
  }
}

// ---------- fused LN2 + MLP (GELU) + residual, in-place on x1 ----------
__launch_bounds__(256, 1)
__global__ void mlp_kernel(const _Float16* __restrict__ w1T,
                           const float* __restrict__ b1,
                           const _Float16* __restrict__ w2T,
                           const float* __restrict__ b2,
                           const float* __restrict__ ln_g,
                           const float* __restrict__ ln_b,
                           float* __restrict__ x1) {
  __shared__ _Float16 s_xln[WS][264];
  __shared__ _Float16 s_h[WS][1040];
  const int tid = threadIdx.x;
  const int wv = __builtin_amdgcn_readfirstlane(tid >> 5);
  const int lane = tid & 31;
  const int nl = lane & 15;
  const int mh = (lane & 16) ? 8 : 0;
  const size_t tok0 = (size_t)blockIdx.x * WS;

#pragma unroll
  for (int r = 0; r < 8; ++r) {
    const int j = wv * 8 + r;
    ln_row_to_lds(x1 + (tok0 + j) * DIM, ln_g, ln_b, &s_xln[j][0]);
  }
  __syncthreads();

  // h = gelu(xln @ w1 + b1): (64x256)@(256x1024), 4 x 64 tiles
  for (int t = wv; t < 256; t += 8) {
    const int mt = t & 3, nt = t >> 2;
    v8f c = {};
#pragma unroll
    for (int kk = 0; kk < 8; ++kk) {
      v16h a = load_a_frag(&s_xln[mt * 16][kk * 32], 264);
      v16h b = load_b_frag(w1T + (size_t)(nt * 16) * DIM + kk * 32, DIM);
      c = wmma16(a, b, c);
    }
    const int n = nt * 16 + nl;
    const int m0 = mt * 16 + mh;
    const float bb = b1[n];
#pragma unroll
    for (int r = 0; r < 8; ++r) {
      const float v = c[r] + bb;
      const float gl = 0.5f * v * (1.0f + erff(v * 0.70710678118654752f));
      s_h[m0 + r][n] = (_Float16)gl;
    }
  }
  __syncthreads();

  // out = h @ w2 + b2 + x1: (64x1024)@(1024x256), 4 x 16 tiles, K = 32 x 32
  for (int t = wv; t < 64; t += 8) {
    const int mt = t & 3, nt = t >> 2;
    v8f c = {};
    for (int kk = 0; kk < 32; ++kk) {
      v16h a = load_a_frag(&s_h[mt * 16][kk * 32], 1040);
      v16h b = load_b_frag(w2T + (size_t)(nt * 16) * MLP_DIM + kk * 32, MLP_DIM);
      c = wmma16(a, b, c);
    }
    const int n = nt * 16 + nl;
    const int m0 = mt * 16 + mh;
    const float bb = b2[n];
#pragma unroll
    for (int r = 0; r < 8; ++r) {
      const size_t idx = (tok0 + m0 + r) * DIM + n;
      x1[idx] = x1[idx] + bb + c[r];
    }
  }
}

extern "C" void kernel_launch(void* const* d_in, const int* in_sizes, int n_in,
                              void* d_out, int out_size, void* d_ws, size_t ws_size,
                              hipStream_t stream) {
  const float* x     = (const float*)d_in[0];
  const float* w_qkv = (const float*)d_in[1];
  const float* w_out = (const float*)d_in[2];
  const float* b_out = (const float*)d_in[3];
  const float* pos   = (const float*)d_in[4];
  const float* ln1_g = (const float*)d_in[5];
  const float* ln1_b = (const float*)d_in[6];
  const float* w1    = (const float*)d_in[7];
  const float* b1    = (const float*)d_in[8];
  const float* w2    = (const float*)d_in[9];
  const float* b2    = (const float*)d_in[10];
  const float* ln2_g = (const float*)d_in[11];
  const float* ln2_b = (const float*)d_in[12];
  float* out = (float*)d_out;
  _Float16* ws = (_Float16*)d_ws;

  prep_weights<<<256, 256, 0, stream>>>(w_qkv, w_out, w1, w2, ws);

  const _Float16* wqkvT = ws;
  const _Float16* woutT = wqkvT + 768 * 256;
  const _Float16* w1T   = woutT + 256 * 256;
  const _Float16* w2T   = w1T + 1024 * 256;

  // 2 batches x 1024 windows; out doubles as the x1 intermediate
  attn_kernel<<<2048, 256, 0, stream>>>(x, wqkvT, woutT, b_out, pos, ln1_g, ln1_b, out);
  // 131072 tokens / 64 per block, in-place on out
  mlp_kernel<<<2048, 256, 0, stream>>>(w1T, b1, w2T, b2, ln2_g, ln2_b, out);
}
